// IndexBackpropagationQuantizer_72241349918908
// MI455X (gfx1250) — compile-verified, matrix operand order checked by reading, not run on hardware
//
#include <hip/hip_runtime.h>
#include <hip/hip_bf16.h>

typedef __bf16 bf16_t;
typedef __bf16 v16bf __attribute__((ext_vector_type(16)));
typedef float  v8f   __attribute__((ext_vector_type(8)));
typedef unsigned int v4u __attribute__((ext_vector_type(4)));
typedef unsigned int v8u __attribute__((ext_vector_type(8)));

#define CB_K 1024           // codebook entries
#define CB_C 256            // codebook dim
#define NPIX 65536          // 16*64*64 pixels
#define TOT_F (16*256*64*64)

// workspace layout (bytes)
static constexpr size_t OFF_ZFLAT = 0;                                   // 65536*256 bf16 = 32 MB
static constexpr size_t OFF_CB    = (size_t)NPIX * CB_C * 2;             // 1024*256 bf16 = 512 KB
static constexpr size_t OFF_AVG   = OFF_CB + (size_t)CB_K * CB_C * 2;    // 1024 f32
static constexpr size_t OFF_HIST  = OFF_AVG + CB_K * 4;                  // 1024 f32
static constexpr size_t OFF_SCAL  = OFF_HIST + CB_K * 4;                 // 2 f32 (sumSq, sumH)

// dynamic-LDS layout for k_main (bytes); aT must sit at 0 (TDM lds_addr)
#define SMEM_AT    0        // A tile: 16 rows x 264 u16 (528 B padded rows) = 8448
#define SMEM_LG    8448     // logits: 16 rows x 1040 bf16 = 33280 -> 41728
#define SMEM_PAVG  41728    // 1024 f32 = 4096 -> 45824
#define SMEM_IDX   45824    // 16 int -> 45888
#define SMEM_SH    45888
#define SMEM_SSQ   45892
#define SMEM_SIZE  45952

// ---------------- init: zero global accumulators ----------------
__global__ void k_zero(float* gAvg, float* gHist, float* gScal) {
  int t = threadIdx.x;
  gAvg[t] = 0.f;
  gHist[t] = 0.f;
  if (t < 2) gScal[t] = 0.f;
}

// ---------------- codebook f32 -> bf16 ----------------
__global__ __launch_bounds__(256) void k_cb(const float* __restrict__ cf,
                                            bf16_t* __restrict__ cb) {
  int i = blockIdx.x * 256 + threadIdx.x;   // grid = 1024 -> 262144 elems
  cb[i] = (bf16_t)cf[i];
}

// ---------------- z [B,C,H,W] -> flat_z [N,C] bf16 (LDS transpose) ----------
__global__ __launch_bounds__(256) void k_zflat(const float* __restrict__ z,
                                               bf16_t* __restrict__ zfl) {
  __shared__ __align__(16) bf16_t t[64 * 256];   // 32 KB: [w][c]
  int bh = blockIdx.x;                            // b*64 + h, grid = 1024
  const float* src = z + (size_t)(bh >> 6) * ((size_t)CB_C * 4096)
                       + (size_t)(bh & 63) * 64;  // z[b][c][h][0]
  for (int i = threadIdx.x; i < 64 * 256; i += 256) {
    int c = i >> 6, w = i & 63;                   // coalesced read along w
    t[w * 256 + c] = (bf16_t)src[(size_t)c * 4096 + w];
  }
  __syncthreads();
  uint4* dst = (uint4*)(zfl + (size_t)bh * (64 * 256));
  const uint4* s4 = (const uint4*)t;
  for (int i = threadIdx.x; i < 2048; i += 256) dst[i] = s4[i];  // coalesced b128
}

// ---------------- main fused kernel ----------------
// grid: 4096 blocks (16 pixels each), block: 256 threads = 8 wave32
__global__ __launch_bounds__(256)
void k_main(const float* __restrict__ z, const float* __restrict__ cbf,
            const bf16_t* __restrict__ zfl, const bf16_t* __restrict__ cbb,
            float* __restrict__ outQ, float* __restrict__ outIdx,
            float* __restrict__ gAvg, float* __restrict__ gHist,
            float* __restrict__ gScal) {
  extern __shared__ char smem[];
  unsigned short* aT   = (unsigned short*)(smem + SMEM_AT);  // row stride 264 u16
  bf16_t*         lg   = (bf16_t*)(smem + SMEM_LG);          // row stride 1040
  float*          pAvg = (float*)(smem + SMEM_PAVG);
  int*            idxRow = (int*)(smem + SMEM_IDX);
  float*          sH   = (float*)(smem + SMEM_SH);
  float*          sSq  = (float*)(smem + SMEM_SSQ);

  const int tid  = threadIdx.x;
  const int lane = tid & 31;
  const int wid  = tid >> 5;
  const int n0   = blockIdx.x * 16;

  // ---- TDM: DMA the A tile (16 rows x 512 B) into LDS with 16 B row padding.
  // D# group0: count=1, lds_addr=0, 57-bit global addr, type=2.
  // D# group1: data_size=4B, pad_enable, pad_interval=128 dw, pad_amount=4 dw,
  //            tensor_dim0=128 dw, tensor_dim0_stride=128 dw, tensor_dim1=1<<20,
  //            tile 128 dw x 16 rows.
  if (wid == 0) {
    unsigned long long ga = (unsigned long long)(const void*)zfl
                          + (unsigned long long)n0 * (CB_C * 2);
    v4u g0 = { 1u,
               (unsigned)SMEM_AT,
               (unsigned)ga,
               (unsigned)((ga >> 32) & 0x01FFFFFFu) | 0x80000000u };
    v8u g1 = { (2u << 16) | (1u << 20) | (6u << 22) | (3u << 25),
               (128u & 0xFFFFu) << 16,
               0u,
               (128u << 16) | ((1u << 20) >> 16),
               16u,
               128u,
               0u,
               0u };
    asm volatile("tensor_load_to_lds %0, %1" :: "s"(g0), "s"(g1) : "memory");
  }

  for (int k = tid; k < CB_K; k += 256) pAvg[k] = 0.f;
  if (tid == 0) { sH[0] = 0.f; sSq[0] = 0.f; }

  // warm L2/WGP$ with this wave's codebook slice (global_prefetch_b8)
  __builtin_prefetch(cbb + (size_t)(wid * 128) * CB_C, 0, 1);

  if (wid == 0) __builtin_amdgcn_s_wait_tensorcnt(0);  // s_wait_tensorcnt
  __syncthreads();

  // ---- WMMA: logits[16 x 1024] = A[16x256] x B[256x1024] (bf16 -> f32 acc) ----
  union F { uint4 u[2]; v16bf v; };
  F afrag[8];
  {
    int row = lane & 15;
    int sel = (lane >> 4) * 8;          // lanes 16-31 hold K+8 half
    #pragma unroll
    for (int kc = 0; kc < 8; ++kc) {
      int koff = kc * 32 + sel;
      const uint4* p = (const uint4*)&aT[row * 264 + koff];
      afrag[kc].u[0] = p[0];            // K = koff..koff+7
      afrag[kc].u[1] = p[2];            // K = koff+16..koff+23
    }
  }
  #pragma unroll 2
  for (int t = 0; t < 8; ++t) {
    int ct   = wid * 8 + t;             // code tile 0..63
    int code = ct * 16 + (lane & 15);
    int sel  = (lane >> 4) * 8;
    const uint4* bbase = (const uint4*)(cbb + (size_t)code * CB_C);
    v8f acc = {};
    #pragma unroll
    for (int kc = 0; kc < 8; ++kc) {
      int koff = kc * 32 + sel;
      F bfrag;
      const uint4* p = bbase + (koff >> 3);
      bfrag.u[0] = p[0];
      bfrag.u[1] = p[2];
      acc = __builtin_amdgcn_wmma_f32_16x16x32_bf16(
          false, afrag[kc].v, false, bfrag.v, (short)0, acc, false, false);
    }
    int M0 = (lane >> 4) << 3;          // lanes 16-31 hold rows 8..15
    int N  = lane & 15;
    #pragma unroll
    for (int v = 0; v < 8; ++v) lg[(M0 + v) * 1040 + ct * 16 + N] = (bf16_t)acc[v];
  }
  __syncthreads();

  // ---- fused softmax(T=0.01)/argmax/entropy per row; wave handles 2 rows ----
  const float invT = 100.0f;
  for (int rr = 0; rr < 2; ++rr) {
    int row = wid * 2 + rr;
    const bf16_t* lrow = lg + row * 1040;
    float m = -1e30f; int mi = 0;
    for (int k = lane; k < CB_K; k += 32) {
      float v = (float)lrow[k];
      if (v > m) { m = v; mi = k; }
    }
    #pragma unroll
    for (int o = 16; o > 0; o >>= 1) {
      float om = __shfl_xor(m, o, 32);
      int  omi = __shfl_xor(mi, o, 32);
      if (om > m || (om == m && omi < mi)) { m = om; mi = omi; }
    }
    float amax = m * invT;
    float sE = 0.f, sAE = 0.f;
    for (int k = lane; k < CB_K; k += 32) {
      float aff = (float)lrow[k] * invT;
      float e = __expf(aff - amax);
      sE += e; sAE += aff * e;
    }
    #pragma unroll
    for (int o = 16; o > 0; o >>= 1) {
      sE  += __shfl_xor(sE, o, 32);
      sAE += __shfl_xor(sAE, o, 32);
    }
    float invS = 1.0f / sE;
    for (int k = lane; k < CB_K; k += 32) {
      float p = __expf((float)lrow[k] * invT - amax) * invS;
      atomicAdd(&pAvg[k], p);           // ds_add_f32
    }
    if (lane == 0) {
      float lse = amax + __logf(sE);
      atomicAdd(sH, lse - sAE * invS);  // sample entropy contribution
      idxRow[row] = mi;
      outIdx[n0 + row] = (float)mi;
      atomicAdd(&gHist[mi], 1.0f);
    }
  }
  __syncthreads();

  // ---- gather quantized output + MSE loss (16 pixels share b,h; w0..w0+15) ----
  int bb = n0 >> 12;
  int hw = n0 & 4095;
  float accSq = 0.f;
  for (int i = tid; i < 4096; i += 256) {
    int c = i >> 4, wl = i & 15;
    int idx = idxRow[wl];
    float q = cbf[idx * CB_C + c];
    size_t goff = ((size_t)(bb * CB_C + c) << 12) + (size_t)(hw + wl);
    float zv = z[goff];
    outQ[goff] = q;
    float d = zv - q;
    accSq += d * d;
  }
  #pragma unroll
  for (int o = 16; o > 0; o >>= 1) accSq += __shfl_xor(accSq, o, 32);
  if (lane == 0) atomicAdd(sSq, accSq);
  __syncthreads();

  for (int k = tid; k < CB_K; k += 256) atomicAdd(&gAvg[k], pAvg[k]);
  if (tid == 0) { atomicAdd(&gScal[0], sSq[0]); atomicAdd(&gScal[1], sH[0]); }
}

// ---------------- finalize scalars ----------------
__global__ __launch_bounds__(1024)
void k_final(const float* __restrict__ gAvg, const float* __restrict__ gHist,
             const float* __restrict__ gScal, float* __restrict__ outScal) {
  __shared__ float s1[1024];
  __shared__ float s2[1024];
  int t = threadIdx.x;
  const float invN = 1.0f / (float)NPIX;
  float ap = gAvg[t] * invN;
  s1[t] = -ap * __logf(ap + 1e-5f);                   // avg_entropy terms
  float pu = gHist[t] * invN;
  s2[t] = -pu * __logf(fmaxf(pu, 1e-10f));            // perplexity exponent terms
  __syncthreads();
  for (int s = 512; s > 0; s >>= 1) {
    if (t < s) { s1[t] += s1[t + s]; s2[t] += s2[t + s]; }
    __syncthreads();
  }
  if (t == 0) {
    float avgE = s1[0];
    float perp = __expf(s2[0]);
    float msq  = gScal[0] / (float)TOT_F;
    float sampleE = gScal[1] * invN;
    outScal[0] = perp;                 // perplexity
    outScal[1] = 2.0f * msq;           // loss_vq (two numerically-equal terms)
    outScal[2] = msq;                  // loss_commit
    outScal[3] = sampleE - avgE;       // loss_entropy
  }
}

extern "C" void kernel_launch(void* const* d_in, const int* in_sizes, int n_in,
                              void* d_out, int out_size, void* d_ws, size_t ws_size,
                              hipStream_t stream) {
  const float* z   = (const float*)d_in[0];   // [16,256,64,64] f32
  const float* cbf = (const float*)d_in[1];   // [1024,256] f32

  char* ws = (char*)d_ws;
  bf16_t* zfl  = (bf16_t*)(ws + OFF_ZFLAT);
  bf16_t* cbb  = (bf16_t*)(ws + OFF_CB);
  float*  gAvg = (float*)(ws + OFF_AVG);
  float*  gHist= (float*)(ws + OFF_HIST);
  float*  gScal= (float*)(ws + OFF_SCAL);

  float* outF    = (float*)d_out;
  float* outQ    = outF;                       // 16,777,216 floats
  float* outIdx  = outF + (size_t)TOT_F;       // 65,536 (indices as f32)
  float* outScal = outIdx + NPIX;              // 4 scalars

  hipLaunchKernelGGL(k_zero,  dim3(1),    dim3(1024), 0, stream, gAvg, gHist, gScal);
  hipLaunchKernelGGL(k_cb,    dim3(1024), dim3(256),  0, stream, cbf, cbb);
  hipLaunchKernelGGL(k_zflat, dim3(1024), dim3(256),  0, stream, z, zfl);
  hipLaunchKernelGGL(k_main,  dim3(4096), dim3(256),  SMEM_SIZE, stream,
                     z, cbf, zfl, cbb, outQ, outIdx, gAvg, gHist, gScal);
  hipLaunchKernelGGL(k_final, dim3(1),    dim3(1024), 0, stream,
                     gAvg, gHist, gScal, outScal);
}
